// ALiBiAttention_86672440033788
// MI455X (gfx1250) — matrix operand order repaired and checked by reading, not compiled, not run on hardware
//
#include <hip/hip_runtime.h>

#define D_MODEL 1024
#define SEQ     2048
#define NH      16
#define DK      64

typedef __attribute__((ext_vector_type(16))) __bf16 v16bf;
typedef __attribute__((ext_vector_type(8)))  __bf16 v8bf;
typedef __attribute__((ext_vector_type(8)))  float  v8f;

__device__ __forceinline__ unsigned short f2bf(float f) {
    unsigned u = __builtin_bit_cast(unsigned, f);
    u += 0x7FFFu + ((u >> 16) & 1u);          // round-to-nearest-even
    return (unsigned short)(u >> 16);
}

__device__ __forceinline__ v16bf cat8(v8bf a, v8bf b) {
    return __builtin_shufflevector(a, b, 0,1,2,3,4,5,6,7,8,9,10,11,12,13,14,15);
}

__device__ __forceinline__ v8f wmma_bf16(v16bf a, v16bf b, v8f c) {
    return __builtin_amdgcn_wmma_f32_16x16x32_bf16(false, a, false, b,
                                                   (short)0, c, false, false);
}

// ---------------------------------------------------------------------------
// One-time f32 -> packed bf16 conversion (bandwidth pass; halves GEMM traffic)
// ---------------------------------------------------------------------------
__global__ __launch_bounds__(256) void cvt_bf16(const float* __restrict__ src,
                                                uint4* __restrict__ dst, int n8)
{
    int i = blockIdx.x * blockDim.x + threadIdx.x;
    if (i >= n8) return;
    const float4* s = (const float4*)src + (size_t)i * 2;
    float4 a = s[0], b = s[1];
    uint4 o;
    o.x = (unsigned)f2bf(a.x) | ((unsigned)f2bf(a.y) << 16);
    o.y = (unsigned)f2bf(a.z) | ((unsigned)f2bf(a.w) << 16);
    o.z = (unsigned)f2bf(b.x) | ((unsigned)f2bf(b.y) << 16);
    o.w = (unsigned)f2bf(b.z) | ((unsigned)f2bf(b.w) << 16);
    dst[i] = o;
}

// ---------------------------------------------------------------------------
// bf16 GEMM: C[M,N] = X[M,K] @ W[N,K]^T + bias, K = N = 1024, M = 4096.
// Double-buffered LDS staged with global_load_async_to_lds_b128 (ASYNCcnt).
// Workgroup (256 thr = 8 waves) computes 128x64; each wave 32x32 (2x2 WMMA).
// mode 0/1: bf16 out at [B,H,S,dk] (Q/K)   mode 2: bf16 at [B,H,dk,S] (V^T)
// mode 3:   f32 row-major [M,N] (final projection)
// ---------------------------------------------------------------------------
#define BM   128
#define BN   64
#define BK   32
#define PADK 40   // halfs per LDS row (80B: 16B-aligned chunks, skews banks)

__device__ __forceinline__ void issue_tile(
    const unsigned short* __restrict__ X, const unsigned short* __restrict__ W,
    unsigned short* As, unsigned short* Bs, int m0, int n0, int k0, int tid)
{
    // A tile: 128 rows x 32 halfs = 512 x 16B chunks; 2 per thread
    #pragma unroll
    for (int ci = 0; ci < 2; ++ci) {
        int c   = tid + ci * 256;
        int row = c >> 2, cc = c & 3;
        unsigned long long ga =
            (unsigned long long)(size_t)(X + (size_t)(m0 + row) * D_MODEL + k0 + cc * 8);
        unsigned la = (unsigned)(size_t)(As + row * PADK + cc * 8);
        asm volatile("global_load_async_to_lds_b128 %0, %1, off"
                     :: "v"(la), "v"(ga) : "memory");
    }
    // B tile: 64 rows x 32 halfs = 256 x 16B chunks; 1 per thread
    {
        int row = tid >> 2, cc = tid & 3;
        unsigned long long ga =
            (unsigned long long)(size_t)(W + (size_t)(n0 + row) * D_MODEL + k0 + cc * 8);
        unsigned la = (unsigned)(size_t)(Bs + row * PADK + cc * 8);
        asm volatile("global_load_async_to_lds_b128 %0, %1, off"
                     :: "v"(la), "v"(ga) : "memory");
    }
}

__global__ __launch_bounds__(256) void qkvo_gemm(
    const unsigned short* __restrict__ X, const unsigned short* __restrict__ W,
    const float* __restrict__ bias, unsigned short* __restrict__ obf,
    float* __restrict__ of32, int mode)
{
    __shared__ __attribute__((aligned(16))) unsigned short As[2][BM * PADK];
    __shared__ __attribute__((aligned(16))) unsigned short Bs[2][BN * PADK];

    const int tid  = threadIdx.x;
    const int lane = tid & 31, wave = tid >> 5;
    const int wm = wave >> 1, wn = wave & 1;
    const int hi = lane >> 4, ln = lane & 15;
    const int m0 = blockIdx.x * BM, n0 = blockIdx.y * BN;

    v8f acc[2][2] = {};

    issue_tile(X, W, &As[0][0], &Bs[0][0], m0, n0, 0, tid);

    constexpr int NIT = D_MODEL / BK;       // 32
    for (int it = 0; it < NIT; ++it) {
        if (it + 1 < NIT) {
            // prefetch next tile into the other buffer (safe: end-barrier of
            // the previous iteration means no wave still reads that buffer)
            issue_tile(X, W, &As[(it + 1) & 1][0], &Bs[(it + 1) & 1][0],
                       m0, n0, (it + 1) * BK, tid);
            // in-order async completion: <=3 outstanding => current tile done
            asm volatile("s_wait_asynccnt 0x3" ::: "memory");
        } else {
            asm volatile("s_wait_asynccnt 0x0" ::: "memory");
        }
        __syncthreads();

        const unsigned short* Ab = &As[it & 1][0];
        const unsigned short* Bb = &Bs[it & 1][0];
        #pragma unroll
        for (int tm = 0; tm < 2; ++tm) {
            const unsigned short* ar = Ab + (wm * 32 + tm * 16 + ln) * PADK;
            v16bf a = cat8(*(const v8bf*)(ar + hi * 8),
                           *(const v8bf*)(ar + 16 + hi * 8));
            #pragma unroll
            for (int tn = 0; tn < 2; ++tn) {
                const unsigned short* br = Bb + (wn * 32 + tn * 16 + ln) * PADK;
                v16bf b = cat8(*(const v8bf*)(br + hi * 8),
                               *(const v8bf*)(br + 16 + hi * 8));
                acc[tm][tn] = wmma_bf16(a, b, acc[tm][tn]);
            }
        }
        __syncthreads();
    }

    // epilogue: bias + scatter per mode. C layout: vgpr r -> row r + hi*8.
    #pragma unroll
    for (int tm = 0; tm < 2; ++tm)
    #pragma unroll
    for (int tn = 0; tn < 2; ++tn) {
        int n = n0 + wn * 32 + tn * 16 + ln;
        float bb = bias[n];
        #pragma unroll
        for (int r = 0; r < 8; ++r) {
            int m = m0 + wm * 32 + tm * 16 + r + hi * 8;
            float val = acc[tm][tn][r] + bb;
            if (mode == 3) {
                of32[(size_t)m * D_MODEL + n] = val;
            } else {
                int bi = m >> 11, s = m & (SEQ - 1);
                int hh = n >> 6,  d = n & (DK - 1);
                size_t idx = (mode == 2)
                    ? (((size_t)(bi * NH + hh) * DK + d) * SEQ + s)
                    : (((size_t)(bi * NH + hh) * SEQ + s) * DK + d);
                obf[idx] = f2bf(val);
            }
        }
    }
}

// ---------------------------------------------------------------------------
// Fused ALiBi attention, streaming softmax. One wave = 32 query rows.
// Q [B,H,S,64] bf16, K [B,H,S,64] bf16, Vt [B,H,64,S] bf16.
// Out bf16 [B,S,H*dk] (feeds the async-bf16 output projection).
// ---------------------------------------------------------------------------
__global__ __launch_bounds__(256) void alibi_attn(
    const unsigned short* __restrict__ Qp,
    const unsigned short* __restrict__ Kp,
    const unsigned short* __restrict__ Vt,
    unsigned short* __restrict__ Ao)
{
    __shared__ __attribute__((aligned(16))) unsigned short P[8 * 32 * PADK];

    const int b = blockIdx.z, h = blockIdx.y;
    const int wave = threadIdx.x >> 5, lane = threadIdx.x & 31;
    const int hi = lane >> 4, ln = lane & 15;
    const int q0 = blockIdx.x * 256 + wave * 32;

    const unsigned short* Qh = Qp + (size_t)(b * NH + h) * SEQ * DK;
    const unsigned short* Kh = Kp + (size_t)(b * NH + h) * SEQ * DK;
    const unsigned short* Vh = Vt + (size_t)(b * NH + h) * DK * SEQ;
    unsigned short* Pw = &P[wave * 32 * PADK];

    const float slope = exp2f((float)h - 8.0f);   // 2^(h-8), faithful to ref

    v16bf aq[2][2];
    #pragma unroll
    for (int tm = 0; tm < 2; ++tm) {
        const unsigned short* qr = Qh + (size_t)(q0 + tm * 16 + ln) * DK;
        #pragma unroll
        for (int ks = 0; ks < 2; ++ks)
            aq[tm][ks] = cat8(*(const v8bf*)(qr + ks * 32 + hi * 8),
                              *(const v8bf*)(qr + ks * 32 + 16 + hi * 8));
    }

    float mrow[2][8], lrow[2][8];
    v8f acc[2][4] = {};
    #pragma unroll
    for (int tm = 0; tm < 2; ++tm)
        #pragma unroll
        for (int r = 0; r < 8; ++r) { mrow[tm][r] = -1e30f; lrow[tm][r] = 0.f; }

    for (int k0 = 0; k0 < SEQ; k0 += 32) {
        if (k0 + 32 < SEQ) {   // global_prefetch_b8 for next chunk
            __builtin_prefetch(Kh + (size_t)(k0 + 32 + lane) * DK, 0, 1);
            __builtin_prefetch(Vh + (size_t)(lane & (DK - 1)) * SEQ + k0 + 32, 0, 1);
        }

        v16bf bk[2][2];
        #pragma unroll
        for (int nt = 0; nt < 2; ++nt) {
            const unsigned short* kr = Kh + (size_t)(k0 + nt * 16 + ln) * DK;
            #pragma unroll
            for (int ks = 0; ks < 2; ++ks)
                bk[nt][ks] = cat8(*(const v8bf*)(kr + ks * 32 + hi * 8),
                                  *(const v8bf*)(kr + ks * 32 + 16 + hi * 8));
        }

        #pragma unroll
        for (int tm = 0; tm < 2; ++tm) {
            v8f sc0 = {}, sc1 = {};
            sc0 = wmma_bf16(aq[tm][0], bk[0][0], sc0);
            sc0 = wmma_bf16(aq[tm][1], bk[0][1], sc0);
            sc1 = wmma_bf16(aq[tm][0], bk[1][0], sc1);
            sc1 = wmma_bf16(aq[tm][1], bk[1][1], sc1);

            float rescale[8];
            #pragma unroll
            for (int r = 0; r < 8; ++r) {
                int gi = q0 + tm * 16 + r + hi * 8;
                float s0 = sc0[r] * 0.125f + slope * (float)(gi - (k0 + ln));
                float s1 = sc1[r] * 0.125f + slope * (float)(gi - (k0 + 16 + ln));
                float v = fmaxf(s0, s1);
                v = fmaxf(v, __shfl_xor(v, 1, 32));
                v = fmaxf(v, __shfl_xor(v, 2, 32));
                v = fmaxf(v, __shfl_xor(v, 4, 32));
                v = fmaxf(v, __shfl_xor(v, 8, 32));
                float nm = fmaxf(mrow[tm][r], v);
                float sc = __expf(mrow[tm][r] - nm);
                mrow[tm][r] = nm;
                rescale[r]  = sc;
                float p0 = __expf(s0 - nm);
                float p1 = __expf(s1 - nm);
                unsigned short* pr = Pw + (tm * 16 + r + hi * 8) * PADK;
                pr[ln]      = f2bf(p0);
                pr[16 + ln] = f2bf(p1);
                float rs = p0 + p1;
                rs += __shfl_xor(rs, 1, 32);
                rs += __shfl_xor(rs, 2, 32);
                rs += __shfl_xor(rs, 4, 32);
                rs += __shfl_xor(rs, 8, 32);
                lrow[tm][r] = lrow[tm][r] * sc + rs;
            }
            #pragma unroll
            for (int dt = 0; dt < 4; ++dt)
                #pragma unroll
                for (int r = 0; r < 8; ++r)
                    acc[tm][dt][r] *= rescale[r];
        }

        // same-wave DS is in-order; wait for P stores before fragment reads
        asm volatile("s_wait_dscnt 0" ::: "memory");

        v16bf bv[4];
        #pragma unroll
        for (int dt = 0; dt < 4; ++dt) {
            const unsigned short* vr = Vh + (size_t)(dt * 16 + ln) * SEQ + k0;
            bv[dt] = cat8(*(const v8bf*)(vr + hi * 8),
                          *(const v8bf*)(vr + 16 + hi * 8));
        }
        #pragma unroll
        for (int tm = 0; tm < 2; ++tm) {
            const unsigned short* pr = Pw + (tm * 16 + ln) * PADK;
            v16bf pa = cat8(*(const v8bf*)(pr + hi * 8),
                            *(const v8bf*)(pr + 16 + hi * 8));
            #pragma unroll
            for (int dt = 0; dt < 4; ++dt)
                acc[tm][dt] = wmma_bf16(pa, bv[dt], acc[tm][dt]);
        }
    }

    // normalize, store out[b, s, h, d] as bf16
    #pragma unroll
    for (int tm = 0; tm < 2; ++tm)
    #pragma unroll
    for (int dt = 0; dt < 4; ++dt)
    #pragma unroll
    for (int r = 0; r < 8; ++r) {
        int s = q0 + tm * 16 + r + hi * 8;
        int d = dt * 16 + ln;
        float o = acc[tm][dt][r] / lrow[tm][r];
        Ao[(((size_t)(b * SEQ + s)) * NH + h) * DK + d] = f2bf(o);
    }
}

// ---------------------------------------------------------------------------
extern "C" void kernel_launch(void* const* d_in, const int* in_sizes, int n_in,
                              void* d_out, int out_size, void* d_ws, size_t ws_size,
                              hipStream_t stream) {
    const float* query = (const float*)d_in[0];
    const float* key_  = (const float*)d_in[1];
    const float* value = (const float*)d_in[2];
    const float* Wq = (const float*)d_in[3];
    const float* bq = (const float*)d_in[4];
    const float* Wk = (const float*)d_in[5];
    const float* bk = (const float*)d_in[6];
    const float* Wv = (const float*)d_in[7];
    const float* bv = (const float*)d_in[8];
    const float* Wo = (const float*)d_in[9];
    const float* bo = (const float*)d_in[10];

    const size_t nX = (size_t)4096 * 1024;   // activation elems (4 Mi)
    const size_t nW = (size_t)1024 * 1024;   // weight elems (1 Mi)

    // workspace layout (bf16 halfs): bf16 inputs/weights, Q/K/V^T, attn out
    unsigned short* xq = (unsigned short*)d_ws;
    unsigned short* xk = xq + nX;
    unsigned short* xv = xk + nX;
    unsigned short* wq = xv + nX;
    unsigned short* wk = wq + nW;
    unsigned short* wv = wk + nW;
    unsigned short* wo = wv + nW;
    unsigned short* qp = wo + nW;
    unsigned short* kp = qp + nX;
    unsigned short* vt = kp + nX;
    unsigned short* ao = vt + nX;

    // 1) one-time f32 -> bf16 conversion passes
    cvt_bf16<<<(int)(nX / 8 / 256), 256, 0, stream>>>(query, (uint4*)xq, (int)(nX / 8));
    cvt_bf16<<<(int)(nX / 8 / 256), 256, 0, stream>>>(key_,  (uint4*)xk, (int)(nX / 8));
    cvt_bf16<<<(int)(nX / 8 / 256), 256, 0, stream>>>(value, (uint4*)xv, (int)(nX / 8));
    cvt_bf16<<<(int)(nW / 8 / 256), 256, 0, stream>>>(Wq, (uint4*)wq, (int)(nW / 8));
    cvt_bf16<<<(int)(nW / 8 / 256), 256, 0, stream>>>(Wk, (uint4*)wk, (int)(nW / 8));
    cvt_bf16<<<(int)(nW / 8 / 256), 256, 0, stream>>>(Wv, (uint4*)wv, (int)(nW / 8));
    cvt_bf16<<<(int)(nW / 8 / 256), 256, 0, stream>>>(Wo, (uint4*)wo, (int)(nW / 8));

    dim3 gg(4096 / BM, D_MODEL / BN);   // 32 x 16 workgroups, M = B*S = 4096

    // 2) QKV projections (async-staged bf16 WMMA GEMMs)
    qkvo_gemm<<<gg, 256, 0, stream>>>(xq, wq, bq, qp, nullptr, 0);
    qkvo_gemm<<<gg, 256, 0, stream>>>(xk, wk, bk, kp, nullptr, 1);
    qkvo_gemm<<<gg, 256, 0, stream>>>(xv, wv, bv, vt, nullptr, 2);

    // 3) fused ALiBi attention (flash-style streaming softmax)
    alibi_attn<<<dim3(SEQ / 256, NH, 2), 256, 0, stream>>>(qp, kp, vt, ao);

    // 4) output projection -> f32 d_out
    qkvo_gemm<<<gg, 256, 0, stream>>>(ao, wo, bo, nullptr, (float*)d_out, 3);
}
